// PairformerBlock_32744830664944
// MI455X (gfx1250) — matrix operand order repaired and checked by reading, not compiled, not run on hardware
//
#include <hip/hip_runtime.h>

// ---------------------------------------------------------------------------
// CDNA5 (gfx1250) Pairformer block. wave32, WMMA bf16 (16x16x32), fp32 accum.
// Tensor Data Mover (TDM) used for bf16 2D tile -> LDS DMA where available.
// ---------------------------------------------------------------------------

typedef __attribute__((ext_vector_type(16))) __bf16 v16bf;
typedef __attribute__((ext_vector_type(8)))  float  v8f;

#define NRES 192
#define NN   (192 * 192)

#define ACT_NONE 0
#define ACT_SIG  1
#define ACT_RELU 2

union AF16 { uint4 q[2]; __bf16 h[16]; v16bf v; };
union F16F { float4 f4[4]; float f[16]; };

__device__ __forceinline__ v8f wmma_bf16(v16bf a, v16bf b, v8f c) {
  return __builtin_amdgcn_wmma_f32_16x16x32_bf16(false, a, false, b, (short)0, c,
                                                 false, false);
}

__device__ __forceinline__ v16bf ld16(const __bf16* p0, const __bf16* p1) {
  AF16 u;
  u.q[0] = *(const uint4*)p0;
  u.q[1] = *(const uint4*)p1;
  return u.v;
}

// A fragment (16x32 bf16) from row-major LDS tile (stride in elements).
// Lane L<16: row L, K {0-7,16-23}; lane>=16: row L-16, K {8-15,24-31}.
__device__ __forceinline__ v16bf afrag(const __bf16* base, int stride, int lane) {
  const __bf16* r = base + (lane & 15) * stride + ((lane < 16) ? 0 : 8);
  return ld16(r, r + 16);
}

// B fragment (32x16 bf16) from Bt[n][k] LDS tile: lane holds col n = lane&15,
// K {0-7,16-23} (lanes 0-15) or K {8-15,24-31} (lanes 16-31).
__device__ __forceinline__ v16bf bfrag(const __bf16* base, int stride, int lane) {
  const __bf16* r = base + (lane & 15) * stride + ((lane < 16) ? 0 : 8);
  return ld16(r, r + 16);
}

// ---------------------------------------------------------------------------
// Tensor Data Mover: DMA a 2D bf16 tile (tile_rows x 32 elems, arbitrary row
// stride) from global into a padded LDS tile laid out as [rows][40] bf16.
// TDM pad: every 16 DWORDs (32 bf16) insert 4 DWORDs (8 bf16) -> [40] rows.
// 6-arg builtin form (clang-23 / therock-10.0): the extra int32x8 group slot
// is zero for 2D tensors. Issue from one wave only; TDM ignores EXEC.
// ---------------------------------------------------------------------------
#if defined(__HIP_DEVICE_COMPILE__) && __has_builtin(__builtin_amdgcn_tensor_load_to_lds)
#define HAVE_TDM 1
typedef __attribute__((ext_vector_type(4))) unsigned int v4u_t;
typedef __attribute__((ext_vector_type(8))) int v8i_t;
typedef __attribute__((ext_vector_type(4))) int v4i_t;

__device__ __forceinline__ void tdm_load_tile_bf16(const __bf16* gsrc,
                                                   unsigned lds_off,
                                                   int tile_rows,
                                                   int row_stride_elems) {
  const unsigned long long ga = (unsigned long long)(uintptr_t)gsrc;
  const unsigned big = 1u << 20;  // tensor dims large enough: tiles never OOB
  v4u_t g0;
  g0.x = 1u;                                     // count=1, user descriptor
  g0.y = lds_off;                                // lds_addr (bytes)
  g0.z = (unsigned)(ga & 0xFFFFFFFFu);           // global_addr[31:0]
  g0.w = (unsigned)((ga >> 32) & 0x01FFFFFFu) |  // global_addr[56:32]
         (2u << 30);                             // type=2 ("image")
  v8i_t g1;
  g1[0] = (int)((1u << 16)      // data_size = 1 -> 2 bytes
              | (1u << 20)      // pad_enable
              | (3u << 22)      // pad_interval: 16 DWORDs
              | (3u << 25));    // pad_amount:   4 DWORDs
  g1[1] = (int)((big & 0xFFFFu) << 16);                 // tensor_dim0 lo16
  g1[2] = (int)((big >> 16) | ((big & 0xFFFFu) << 16)); // dim0 hi | dim1 lo
  g1[3] = (int)((big >> 16) | (32u << 16));             // dim1 hi | tile_dim0=32
  g1[4] = tile_rows;                                    // tile_dim1 (dim2=0)
  g1[5] = row_stride_elems;                             // tensor_dim0_stride lo32
  g1[6] = 0;
  g1[7] = 0;
  const v4i_t z4 = {0, 0, 0, 0};
  const v8i_t z8 = {0, 0, 0, 0, 0, 0, 0, 0};
  __builtin_amdgcn_tensor_load_to_lds(g0, g1, z4, z4, z8, 0);
}
#else
#define HAVE_TDM 0
#endif

// ---------------------------------------------------------------------------
// Generic tiled GEMM: C[M,N] = epilogue(A[M,K] @ W[K,N] + bias [+ resid])
// Tile 64x64, BK=32, 128 threads = 4 waves, each wave a 32x32 tile (2x2 WMMA).
// M % 64 == 0, K % 32 == 0 in all uses; N handled with bounds (N=16 case).
// bf16 A tiles are DMA'd by the TDM; f32 A tiles converted through VGPRs.
// ---------------------------------------------------------------------------
template <typename TA, typename TO, int ACT, bool RESID>
__global__ __launch_bounds__(128) void gemm_kernel(
    const TA* __restrict__ A, const float* __restrict__ W,
    const float* __restrict__ bias, const float* __restrict__ resid,
    TO* __restrict__ C, int M, int N, int K) {
  __shared__ __align__(16) __bf16 As[64][40];
  __shared__ __align__(16) __bf16 Bs[64][40];
  const int tid = threadIdx.x, lane = tid & 31, wave = tid >> 5;
  const int n0 = blockIdx.x * 64, m0 = blockIdx.y * 64;
  const int wr = (wave >> 1) * 32, wc = (wave & 1) * 32;
  v8f acc[2][2] = {};
  const int arow = tid >> 1, ahalf = (tid & 1) * 16;

  for (int k0 = 0; k0 < K; k0 += 32) {
    // ---- A tile (64x32) -> bf16 LDS
    auto load_a_vgpr = [&]() {
      const TA* src = A + (size_t)(m0 + arow) * K + k0 + ahalf;
      __builtin_prefetch((const void*)(src + 32), 0, 1);  // global_prefetch_b8
      AF16 u;
      if constexpr (sizeof(TA) == 4) {
        F16F t;
        t.f4[0] = *(const float4*)(src + 0);
        t.f4[1] = *(const float4*)(src + 4);
        t.f4[2] = *(const float4*)(src + 8);
        t.f4[3] = *(const float4*)(src + 12);
#pragma unroll
        for (int i = 0; i < 16; ++i) u.h[i] = (__bf16)t.f[i];
      } else {
        u.q[0] = *(const uint4*)(src);
        u.q[1] = *(const uint4*)(src + 8);
      }
      *(uint4*)&As[arow][ahalf] = u.q[0];
      *(uint4*)&As[arow][ahalf + 8] = u.q[1];
    };
#if HAVE_TDM
    if constexpr (sizeof(TA) == 2) {
      if (wave == 0) {
        tdm_load_tile_bf16((const __bf16*)A + (size_t)m0 * K + k0,
                           (unsigned)(uintptr_t)&As[0][0], 64, K);
        __builtin_amdgcn_s_wait_tensorcnt(0);
      }
    } else {
      load_a_vgpr();
    }
#else
    load_a_vgpr();
#endif
    // ---- W tile (32x64) -> transposed bf16 LDS Bs[n][k]
    {
      const int nb = (tid & 15) * 4;
      const int kb = tid >> 4;  // 0..7
#pragma unroll
      for (int kr = 0; kr < 4; ++kr) {
        const int kk = kb + kr * 8;
        const float* wrow = W + (size_t)(k0 + kk) * N + n0;
#pragma unroll
        for (int j = 0; j < 4; ++j) {
          const int nn = nb + j;
          const float v = (n0 + nn < N) ? wrow[nn] : 0.f;
          Bs[nn][kk] = (__bf16)v;
        }
      }
    }
    __syncthreads();
    const v16bf a0 = afrag(&As[wr][0], 40, lane);
    const v16bf a1 = afrag(&As[wr + 16][0], 40, lane);
    const v16bf b0 = bfrag(&Bs[wc][0], 40, lane);
    const v16bf b1 = bfrag(&Bs[wc + 16][0], 40, lane);
    acc[0][0] = wmma_bf16(a0, b0, acc[0][0]);
    acc[0][1] = wmma_bf16(a0, b1, acc[0][1]);
    acc[1][0] = wmma_bf16(a1, b0, acc[1][0]);
    acc[1][1] = wmma_bf16(a1, b1, acc[1][1]);
    __syncthreads();
  }
  // ---- epilogue
#pragma unroll
  for (int tm = 0; tm < 2; ++tm)
#pragma unroll
    for (int tn = 0; tn < 2; ++tn) {
      const int nl = wc + tn * 16 + (lane & 15);
      const int n = n0 + nl;
      if (n < N) {
        const float bv = bias ? bias[n] : 0.f;
#pragma unroll
        for (int e = 0; e < 8; ++e) {
          const int m = m0 + wr + tm * 16 + ((lane < 16) ? e : e + 8);
          float v = acc[tm][tn][e] + bv;
          if (ACT == ACT_SIG)  v = 1.f / (1.f + __expf(-v));
          if (ACT == ACT_RELU) v = fmaxf(v, 0.f);
          if (RESID)           v += resid[(size_t)m * N + n];
          C[(size_t)m * N + n] = (TO)v;
        }
      }
    }
}

// ---------------------------------------------------------------------------
// [i,k,c] (bf16) -> channel-major [c][i][k] (bf16); swap!=0 reads t[k,i,c]
// (turns both tri-mul einsum variants into the same row-major GEMM form).
// grid (192, 6), block 256.
// ---------------------------------------------------------------------------
__global__ __launch_bounds__(256) void chtrans_kernel(const __bf16* __restrict__ src,
                                                      __bf16* __restrict__ dst,
                                                      int swap) {
  __shared__ __bf16 tile[32][128];
  const int i = blockIdx.x, k0 = blockIdx.y * 32;
  for (int e = threadIdx.x; e < 32 * 128; e += 256) {
    const int kk = e >> 7, cc = e & 127;
    const size_t row = swap ? ((size_t)(k0 + kk) * NRES + i)
                            : ((size_t)i * NRES + (k0 + kk));
    tile[kk][cc] = src[row * 128 + cc];
  }
  __syncthreads();
  for (int e = threadIdx.x; e < 32 * 128; e += 256) {
    const int cc = e >> 5, kk = e & 31;
    dst[(size_t)cc * NN + (size_t)i * NRES + k0 + kk] = tile[kk][cc];
  }
}

// ---------------------------------------------------------------------------
// Triangular-multiplicative einsum: U[i,j,c] = sum_k G[c][i][k] * X[c][j][k]
// 128 independent 192x192x192 bf16 GEMMs. grid (3,3,128), block 128 (4 waves).
// Both operand tiles are pure 2D bf16 strided tiles -> TDM DMA into LDS.
// ---------------------------------------------------------------------------
__global__ __launch_bounds__(128) void tri_einsum_kernel(
    const __bf16* __restrict__ G, const __bf16* __restrict__ X,
    __bf16* __restrict__ U) {
  __shared__ __align__(16) __bf16 As[64][40];
  __shared__ __align__(16) __bf16 Bs[64][40];
  const int c = blockIdx.z;
  const int i0 = blockIdx.y * 64, j0 = blockIdx.x * 64;
  const __bf16* Gc = G + (size_t)c * NN;
  const __bf16* Xc = X + (size_t)c * NN;
  const int tid = threadIdx.x, lane = tid & 31, wave = tid >> 5;
  const int wr = (wave >> 1) * 32, wc = (wave & 1) * 32;
  const int arow = tid >> 1, ahalf = (tid & 1) * 16;
  v8f acc[2][2] = {};

  for (int k0 = 0; k0 < NRES; k0 += 32) {
#if HAVE_TDM
    if (wave == 0) {
      tdm_load_tile_bf16(Gc + (size_t)i0 * NRES + k0,
                         (unsigned)(uintptr_t)&As[0][0], 64, NRES);
      tdm_load_tile_bf16(Xc + (size_t)j0 * NRES + k0,
                         (unsigned)(uintptr_t)&Bs[0][0], 64, NRES);
      __builtin_amdgcn_s_wait_tensorcnt(0);
    }
#else
    {
      const __bf16* ga = Gc + (size_t)(i0 + arow) * NRES + k0 + ahalf;
      const __bf16* xa = Xc + (size_t)(j0 + arow) * NRES + k0 + ahalf;
      *(uint4*)&As[arow][ahalf]     = *(const uint4*)(ga);
      *(uint4*)&As[arow][ahalf + 8] = *(const uint4*)(ga + 8);
      *(uint4*)&Bs[arow][ahalf]     = *(const uint4*)(xa);
      *(uint4*)&Bs[arow][ahalf + 8] = *(const uint4*)(xa + 8);
    }
#endif
    __syncthreads();
    const v16bf a0 = afrag(&As[wr][0], 40, lane);
    const v16bf a1 = afrag(&As[wr + 16][0], 40, lane);
    const v16bf b0 = bfrag(&Bs[wc][0], 40, lane);
    const v16bf b1 = bfrag(&Bs[wc + 16][0], 40, lane);
    acc[0][0] = wmma_bf16(a0, b0, acc[0][0]);
    acc[0][1] = wmma_bf16(a0, b1, acc[0][1]);
    acc[1][0] = wmma_bf16(a1, b0, acc[1][0]);
    acc[1][1] = wmma_bf16(a1, b1, acc[1][1]);
    __syncthreads();
  }
#pragma unroll
  for (int tm = 0; tm < 2; ++tm)
#pragma unroll
    for (int tn = 0; tn < 2; ++tn) {
      const int n = j0 + wc + tn * 16 + (lane & 15);
#pragma unroll
      for (int e = 0; e < 8; ++e) {
        const int m = i0 + wr + tm * 16 + ((lane < 16) ? e : e + 8);
        U[((size_t)m * NRES + n) * 128 + c] = (__bf16)acc[tm][tn][e];
      }
    }
}

// ---------------------------------------------------------------------------
// Fused triangular attention slice. grid (192 slices, 16 heads), block 384
// (12 waves x 16 query rows). starting!=0: slice=i, rows=j (keys over k of
// pair[i,:,:]); else slice=j, rows=i (keys over k of pair[:,j,:]).
// Each wave keeps its 16 pair rows as 4 A-fragments in registers, reused for
// Q/K/V projections; flash-style online softmax over causal key chunks of 32.
// Output O[rowGlobal, h*32+c] in bf16 (summed over heads by a later GEMM).
// ---------------------------------------------------------------------------
__global__ __launch_bounds__(384) void tri_attn_kernel(
    const float* __restrict__ pair,
    const float* __restrict__ Wq, const float* __restrict__ bq,
    const float* __restrict__ Wk, const float* __restrict__ bk,
    const float* __restrict__ Wv, const float* __restrict__ bv,
    __bf16* __restrict__ Ows, int starting) {
  __shared__ __align__(16) __bf16 WqT[32][32];
  __shared__ __align__(16) __bf16 WkT[32][32];
  __shared__ __align__(16) __bf16 WvT[32][32];
  __shared__ __align__(16) __bf16 Qh[192][32];
  __shared__ __align__(16) __bf16 Kh[192][32];
  __shared__ __align__(16) __bf16 VT[32][192];
  __shared__ __align__(16) __bf16 Pst[12][16][32];

  const int slice = blockIdx.x, h = blockIdx.y;
  const int tid = threadIdx.x, lane = tid & 31, wave = tid >> 5;
  const int m0 = wave * 16;
  const float scale = 0.17677669529663687f;  // 1/sqrt(32)

  // ---- phase A: this wave's 16 pair rows as 4 A-fragments (registers)
  const int rloc = m0 + (lane & 15);
  const int koff = (lane < 16) ? 0 : 8;
  const float* rowp =
      pair + (size_t)(starting ? slice * NRES + rloc : rloc * NRES + slice) * 128;
  v16bf afr[4];
#pragma unroll
  for (int s = 0; s < 4; ++s) {
    F16F t;
    t.f4[0] = *(const float4*)(rowp + s * 32 + koff);
    t.f4[1] = *(const float4*)(rowp + s * 32 + koff + 4);
    t.f4[2] = *(const float4*)(rowp + s * 32 + koff + 16);
    t.f4[3] = *(const float4*)(rowp + s * 32 + koff + 20);
    AF16 u;
#pragma unroll
    for (int i = 0; i < 16; ++i) u.h[i] = (__bf16)t.f[i];
    afr[s] = u.v;
  }

  // ---- phase B: Q/K/V strips (16x32 each) via WMMA, K-loop over 4 steps
  v8f cQ[2] = {}, cK[2] = {}, cV[2] = {};
  for (int s = 0; s < 4; ++s) {
    for (int e = tid; e < 1024; e += 384) {
      const int n = e & 31, kk = e >> 5;
      const int col = h * 32 + n;
      WqT[n][kk] = (__bf16)Wq[(size_t)(s * 32 + kk) * 512 + col];
      WkT[n][kk] = (__bf16)Wk[(size_t)(s * 32 + kk) * 512 + col];
      WvT[n][kk] = (__bf16)Wv[(size_t)(s * 32 + kk) * 512 + col];
    }
    __syncthreads();
#pragma unroll
    for (int t = 0; t < 2; ++t) {
      cQ[t] = wmma_bf16(afr[s], bfrag(&WqT[t * 16][0], 32, lane), cQ[t]);
      cK[t] = wmma_bf16(afr[s], bfrag(&WkT[t * 16][0], 32, lane), cK[t]);
      cV[t] = wmma_bf16(afr[s], bfrag(&WvT[t * 16][0], 32, lane), cV[t]);
    }
    __syncthreads();
  }
#pragma unroll
  for (int t = 0; t < 2; ++t) {
    const int nl = t * 16 + (lane & 15);
    const float bqv = bq[h * 32 + nl], bkv = bk[h * 32 + nl], bvv = bv[h * 32 + nl];
#pragma unroll
    for (int e = 0; e < 8; ++e) {
      const int row = m0 + ((lane < 16) ? e : e + 8);
      Qh[row][nl] = (__bf16)(cQ[t][e] + bqv);
      Kh[row][nl] = (__bf16)(cK[t][e] + bkv);
      VT[nl][row] = (__bf16)(cV[t][e] + bvv);  // transposed for PV B-fragments
    }
  }
  __syncthreads();

  // ---- phase C: causal online-softmax attention over key chunks of 32
  const v16bf aQ = afrag(&Qh[m0][0], 32, lane);
  float m_run[8], l_run[8];
#pragma unroll
  for (int e = 0; e < 8; ++e) { m_run[e] = -1e9f; l_run[e] = 0.f; }
  v8f cO[2] = {};
  const int nch = (m0 + 47) >> 5;  // chunks covering keys <= m0+15

  for (int kc = 0; kc < nch; ++kc) {
    const int k0 = kc * 32;
    const v16bf bK0 = bfrag(&Kh[k0][0], 32, lane);
    const v16bf bK1 = bfrag(&Kh[k0 + 16][0], 32, lane);
    const v8f z = {};
    const v8f s0 = wmma_bf16(aQ, bK0, z);
    const v8f s1 = wmma_bf16(aQ, bK1, z);
    const int key0 = k0 + (lane & 15);
#pragma unroll
    for (int e = 0; e < 8; ++e) {
      const int rl = (lane < 16) ? e : e + 8;
      const int row = m0 + rl;
      float f0 = (key0 <= row) ? s0[e] * scale : -1e9f;
      float f1 = (key0 + 16 <= row) ? s1[e] * scale : -1e9f;
      float mt = fmaxf(f0, f1);
      mt = fmaxf(mt, __shfl_xor(mt, 1));
      mt = fmaxf(mt, __shfl_xor(mt, 2));
      mt = fmaxf(mt, __shfl_xor(mt, 4));
      mt = fmaxf(mt, __shfl_xor(mt, 8));  // row lives in a 16-lane half
      const float mnew = fmaxf(m_run[e], mt);
      const float corr = __expf(m_run[e] - mnew);
      const float p0 = __expf(f0 - mnew);
      const float p1 = __expf(f1 - mnew);
      float rs = p0 + p1;
      rs += __shfl_xor(rs, 1);
      rs += __shfl_xor(rs, 2);
      rs += __shfl_xor(rs, 4);
      rs += __shfl_xor(rs, 8);
      l_run[e] = l_run[e] * corr + rs;
      m_run[e] = mnew;
      cO[0][e] = cO[0][e] * corr;
      cO[1][e] = cO[1][e] * corr;
      Pst[wave][rl][lane & 15] = (__bf16)p0;
      Pst[wave][rl][16 + (lane & 15)] = (__bf16)p1;
    }
    // wave-local LDS round-trip to reshape P into A-fragment layout
    const v16bf aP = afrag(&Pst[wave][0][0], 32, lane);
#pragma unroll
    for (int t = 0; t < 2; ++t) {
      const int n = t * 16 + (lane & 15);
      const __bf16* p = &VT[n][k0 + ((lane < 16) ? 0 : 8)];
      cO[t] = wmma_bf16(aP, ld16(p, p + 16), cO[t]);
    }
  }

  // ---- write normalized O (bf16)
#pragma unroll
  for (int t = 0; t < 2; ++t)
#pragma unroll
    for (int e = 0; e < 8; ++e) {
      const int row = m0 + ((lane < 16) ? e : e + 8);
      const float ov = cO[t][e] / l_run[e];
      const size_t gr = starting ? (size_t)slice * NRES + row
                                 : (size_t)row * NRES + slice;
      Ows[gr * 512 + h * 32 + t * 16 + (lane & 15)] = (__bf16)ov;
    }
}

// ---------------------------------------------------------------------------
// Pair->single attention (tiny: 192 queries x 192 keys x 16 heads).
// grid (192, 16), block 192. a = (q.k + bias)*scale, softmax over keys.
// ---------------------------------------------------------------------------
__global__ __launch_bounds__(192) void p2s_attn_kernel(
    const float* __restrict__ q, const float* __restrict__ k,
    const float* __restrict__ v, const float* __restrict__ bias,
    __bf16* __restrict__ O2) {
  __shared__ float sc[192];
  __shared__ float pr[192];
  const int i = blockIdx.x, h = blockIdx.y, j = threadIdx.x;
  const float* qp = q + (size_t)i * 512 + h * 32;
  const float* kp = k + (size_t)j * 512 + h * 32;
  float s = 0.f;
#pragma unroll
  for (int c = 0; c < 32; ++c) s += qp[c] * kp[c];
  s = (s + bias[((size_t)i * NRES + j) * 16 + h]) * 0.17677669529663687f;
  sc[j] = s;
  __syncthreads();
  float m = -1e30f;
  for (int jj = 0; jj < NRES; ++jj) m = fmaxf(m, sc[jj]);
  pr[j] = __expf(s - m);
  __syncthreads();
  float sum = 0.f;
  for (int jj = 0; jj < NRES; ++jj) sum += pr[jj];
  if (j < 32) {
    float acc = 0.f;
    for (int jj = 0; jj < NRES; ++jj) acc += pr[jj] * v[(size_t)jj * 512 + h * 32 + j];
    O2[(size_t)i * 512 + h * 32 + j] = (__bf16)(acc / sum);
  }
}

// ---------------------------------------------------------------------------
// Host orchestration
// ---------------------------------------------------------------------------
extern "C" void kernel_launch(void* const* d_in, const int* in_sizes, int n_in,
                              void* d_out, int out_size, void* d_ws, size_t ws_size,
                              hipStream_t stream) {
  const float* single_in = (const float*)d_in[0];
  const float* pair_in   = (const float*)d_in[1];
  // params flattened in dict insertion order, each lin = (W, b):
  // 0 out_mul_a 1 out_mul_b 2 out_mul_o 3 in_mul_a 4 in_mul_b 5 in_mul_o
  // 6 start_q 7 start_k 8 start_v 9 start_o 10 end_q 11 end_k 12 end_v 13 end_o
  // 14 pair_tr_1 15 pair_tr_2 16 p2s_q 17 p2s_k 18 p2s_v 19 p2s_bias 20 p2s_o
  // 21 single_tr_1 22 single_tr_2
  auto Wp = [&](int j) { return (const float*)d_in[2 + 2 * j]; };
  auto Bp = [&](int j) { return (const float*)d_in[3 + 2 * j]; };

  char* ws = (char*)d_ws;
  size_t off = 0;
  auto alloc = [&](size_t bytes) -> void* {
    void* p = ws + off;
    off += (bytes + 255) & ~(size_t)255;
    return p;
  };
  float*  PairA = (float*)alloc((size_t)NN * 128 * 4);
  __bf16* Gb    = (__bf16*)alloc((size_t)NN * 128 * 2);
  __bf16* Xb    = (__bf16*)alloc((size_t)NN * 128 * 2);
  __bf16* chG   = (__bf16*)alloc((size_t)NN * 128 * 2);
  __bf16* chX   = (__bf16*)alloc((size_t)NN * 128 * 2);
  __bf16* Ub    = (__bf16*)alloc((size_t)NN * 128 * 2);
  __bf16* Ows   = (__bf16*)alloc((size_t)NN * 512 * 2);  // also reused as pair_tr hidden
  float*  Bws   = (float*)alloc((size_t)NN * 16 * 4);
  float*  qws   = (float*)alloc((size_t)NRES * 512 * 4);
  float*  kws   = (float*)alloc((size_t)NRES * 512 * 4);
  float*  vws   = (float*)alloc((size_t)NRES * 512 * 4);
  __bf16* O2    = (__bf16*)alloc((size_t)NRES * 512 * 2);
  __bf16* T1    = (__bf16*)alloc((size_t)NRES * 1536 * 2);

  float* out_single = (float*)d_out;
  float* out_pair   = out_single + (size_t)NRES * 384;

  auto gg = [](int M, int N) { return dim3((unsigned)((N + 63) / 64),
                                           (unsigned)((M + 63) / 64)); };
  const dim3 gb(128);

  // ---- tri-mul outgoing: pair -> PairA
  gemm_kernel<float, __bf16, ACT_SIG, false><<<gg(NN, 128), gb, 0, stream>>>(
      pair_in, Wp(0), Bp(0), nullptr, Gb, NN, 128, 128);
  gemm_kernel<float, __bf16, ACT_NONE, false><<<gg(NN, 128), gb, 0, stream>>>(
      pair_in, Wp(1), Bp(1), nullptr, Xb, NN, 128, 128);
  chtrans_kernel<<<dim3(192, 6), 256, 0, stream>>>(Gb, chG, 0);
  chtrans_kernel<<<dim3(192, 6), 256, 0, stream>>>(Xb, chX, 0);
  tri_einsum_kernel<<<dim3(3, 3, 128), 128, 0, stream>>>(chG, chX, Ub);
  gemm_kernel<__bf16, float, ACT_NONE, true><<<gg(NN, 128), gb, 0, stream>>>(
      Ub, Wp(2), Bp(2), pair_in, PairA, NN, 128, 128);

  // ---- tri-mul incoming: PairA -> out_pair
  gemm_kernel<float, __bf16, ACT_SIG, false><<<gg(NN, 128), gb, 0, stream>>>(
      PairA, Wp(3), Bp(3), nullptr, Gb, NN, 128, 128);
  gemm_kernel<float, __bf16, ACT_NONE, false><<<gg(NN, 128), gb, 0, stream>>>(
      PairA, Wp(4), Bp(4), nullptr, Xb, NN, 128, 128);
  chtrans_kernel<<<dim3(192, 6), 256, 0, stream>>>(Gb, chG, 1);
  chtrans_kernel<<<dim3(192, 6), 256, 0, stream>>>(Xb, chX, 1);
  tri_einsum_kernel<<<dim3(3, 3, 128), 128, 0, stream>>>(chG, chX, Ub);
  gemm_kernel<__bf16, float, ACT_NONE, true><<<gg(NN, 128), gb, 0, stream>>>(
      Ub, Wp(5), Bp(5), PairA, out_pair, NN, 128, 128);

  // ---- triangular attention (starting): out_pair -> PairA
  tri_attn_kernel<<<dim3(192, 16), 384, 0, stream>>>(
      out_pair, Wp(6), Bp(6), Wp(7), Bp(7), Wp(8), Bp(8), Ows, 1);
  gemm_kernel<__bf16, float, ACT_NONE, true><<<gg(NN, 128), gb, 0, stream>>>(
      Ows, Wp(9), Bp(9), out_pair, PairA, NN, 128, 512);

  // ---- triangular attention (ending): PairA -> out_pair
  tri_attn_kernel<<<dim3(192, 16), 384, 0, stream>>>(
      PairA, Wp(10), Bp(10), Wp(11), Bp(11), Wp(12), Bp(12), Ows, 0);
  gemm_kernel<__bf16, float, ACT_NONE, true><<<gg(NN, 128), gb, 0, stream>>>(
      Ows, Wp(13), Bp(13), PairA, out_pair, NN, 128, 512);

  // ---- pair transition (in place on out_pair; hidden reuses Ows buffer)
  gemm_kernel<float, __bf16, ACT_RELU, false><<<gg(NN, 512), gb, 0, stream>>>(
      out_pair, Wp(14), Bp(14), nullptr, Ows, NN, 512, 128);
  gemm_kernel<__bf16, float, ACT_NONE, true><<<gg(NN, 128), gb, 0, stream>>>(
      Ows, Wp(15), Bp(15), out_pair, out_pair, NN, 128, 512);

  // ---- pair -> single attention
  gemm_kernel<float, float, ACT_NONE, false><<<gg(NN, 16), gb, 0, stream>>>(
      out_pair, Wp(19), Bp(19), nullptr, Bws, NN, 16, 128);
  gemm_kernel<float, float, ACT_NONE, false><<<gg(NRES, 512), gb, 0, stream>>>(
      single_in, Wp(16), Bp(16), nullptr, qws, NRES, 512, 384);
  gemm_kernel<float, float, ACT_NONE, false><<<gg(NRES, 512), gb, 0, stream>>>(
      single_in, Wp(17), Bp(17), nullptr, kws, NRES, 512, 384);
  gemm_kernel<float, float, ACT_NONE, false><<<gg(NRES, 512), gb, 0, stream>>>(
      single_in, Wp(18), Bp(18), nullptr, vws, NRES, 512, 384);
  p2s_attn_kernel<<<dim3(192, 16), 192, 0, stream>>>(qws, kws, vws, Bws, O2);
  gemm_kernel<__bf16, float, ACT_NONE, true><<<gg(NRES, 384), gb, 0, stream>>>(
      O2, Wp(20), Bp(20), single_in, out_single, NRES, 384, 512);

  // ---- single transition (in place on out_single)
  gemm_kernel<float, __bf16, ACT_RELU, false><<<gg(NRES, 1536), gb, 0, stream>>>(
      out_single, Wp(21), Bp(21), nullptr, T1, NRES, 1536, 384);
  gemm_kernel<__bf16, float, ACT_NONE, true><<<gg(NRES, 384), gb, 0, stream>>>(
      T1, Wp(22), Bp(22), out_single, out_single, NRES, 384, 1536);
}
